// QKNormedAttention_62706522522378
// MI455X (gfx1250) — matrix operand order repaired and checked by reading, not compiled, not run on hardware
//
#include <hip/hip_runtime.h>

typedef __attribute__((ext_vector_type(16))) __bf16 v16bf;
typedef __attribute__((ext_vector_type(8)))  __bf16 v8bf;
typedef __attribute__((ext_vector_type(8)))  float  v8f;

#define N_TOK 4096
#define DIM   1024
#define NHEAD 16
#define HDIM  64

#if defined(__gfx1250__) && __has_builtin(__builtin_amdgcn_tensor_load_to_lds) && \
    __has_builtin(__builtin_amdgcn_s_wait_tensorcnt)
#define ATTN_USE_TDM 1
#else
#define ATTN_USE_TDM 0
#endif

// ---------- helpers ----------

// f32 -> bf16, round-to-nearest-even (bit manipulation; no __bf16 arithmetic needed)
static __device__ __forceinline__ __bf16 f2b(float f) {
  unsigned int x = __builtin_bit_cast(unsigned int, f);
  x += 0x7fffu + ((x >> 16) & 1u);
  unsigned short s = (unsigned short)(x >> 16);
  return __builtin_bit_cast(__bf16, s);
}

// Load a 16x32 bf16 A-fragment (M=16 x K=32) from a row-major matrix.
// CDNA5 16-bit A layout: lane L holds row (L&15); elems 0-7 = K 8*half+0..7,
// elems 8-15 = K 16+8*half+0..7, half = L>>4.
static __device__ __forceinline__ v16bf load_a_frag(const __bf16* __restrict__ src,
                                                    int ld, int row_base, int k0) {
  int lane = threadIdx.x;
  int row  = row_base + (lane & 15);
  int half = (lane >> 4) & 1;
  const __bf16* p = src + (size_t)row * ld + k0 + 8 * half;
  v8bf lo = *(const v8bf*)p;
  v8bf hi = *(const v8bf*)(p + 16);
  v16bf a;
#pragma unroll
  for (int i = 0; i < 8; ++i) { a[i] = lo[i]; a[i + 8] = hi[i]; }
  return a;
}

// Load a 32x16 bf16 B-fragment (K=32 x N=16) from row-major [K x N] storage.
// CDNA5 16-bit B layout: lane L holds column (L&15); elems 0-7 = K 8*half+0..7,
// elems 8-15 = K 16+8*half+0..7, half = L>>4.
static __device__ __forceinline__ v16bf load_b_frag(const __bf16* __restrict__ src,
                                                    int ld, int k0, int n0) {
  int lane = threadIdx.x;
  int n = n0 + (lane & 15);
  int half = (lane >> 4) & 1;
  v16bf b;
#pragma unroll
  for (int s = 0; s < 8; ++s) {
    b[s]     = src[(size_t)(k0 + 8 * half + s) * ld + n];
    b[s + 8] = src[(size_t)(k0 + 16 + 8 * half + s) * ld + n];
  }
  return b;
}

#if ATTN_USE_TDM
typedef __attribute__((ext_vector_type(4))) unsigned int v4u;
typedef __attribute__((ext_vector_type(4))) int          v4i;
typedef __attribute__((ext_vector_type(8))) int          v8i;

// Issue a TDM 2-D tile load (bf16 elements) global -> LDS, per CDNA5 ISA D# layout.
// tile_w/tile_h in elements; tensor_w/tensor_h = full tensor dims; row_stride in elements.
static __device__ __forceinline__ void tdm_load_tile_2d(
    const void* lds_dst, const void* gsrc,
    unsigned tile_w, unsigned tile_h,
    unsigned tensor_w, unsigned tensor_h, unsigned row_stride) {
  unsigned long long ga = (unsigned long long)(uintptr_t)gsrc;
  unsigned lds_off = (unsigned)(uintptr_t)lds_dst;  // low 32 bits of generic LDS ptr = LDS offset
  v4u g0;
  g0[0] = 1u;                                               // count=1, user descriptor
  g0[1] = lds_off;                                          // lds_addr
  g0[2] = (unsigned)ga;                                     // global_addr[31:0]
  g0[3] = ((unsigned)(ga >> 32) & 0x01FFFFFFu) | 0x80000000u; // addr[56:32] | type=2
  v8i g1;
  g1[0] = (int)0x00010000u;                                 // workgroup_mask=0, data_size=1 (2B)
  g1[1] = (int)((tensor_w & 0xFFFFu) << 16);                // tensor_dim0[15:0]
  g1[2] = (int)((tensor_w >> 16) | ((tensor_h & 0xFFFFu) << 16)); // dim0 hi | dim1 lo
  g1[3] = (int)((tensor_h >> 16) | (tile_w << 16));         // dim1 hi | tile_dim0
  g1[4] = (int)tile_h;                                      // tile_dim1, tile_dim2=0
  g1[5] = (int)row_stride;                                  // tensor_dim0_stride[31:0]
  g1[6] = 0;
  g1[7] = 0;
  v4i z4 = {0, 0, 0, 0};
#if __clang_major__ >= 23
  v8i z8 = {0, 0, 0, 0, 0, 0, 0, 0};
  __builtin_amdgcn_tensor_load_to_lds(g0, g1, z4, z4, z8, 0);
#else
  __builtin_amdgcn_tensor_load_to_lds(g0, g1, z4, z4, 0);
#endif
}
#endif  // ATTN_USE_TDM

// ---------- elementwise f32 -> bf16 ----------
__global__ void cvt_bf16_kernel(const float* __restrict__ in, __bf16* __restrict__ out, int n) {
  int i = blockIdx.x * blockDim.x + threadIdx.x;
  if (i < n) out[i] = f2b(in[i]);
}

// ---------- W[n][k] (f32) -> Wt[k][n] (bf16), LDS-tiled transpose ----------
__global__ __launch_bounds__(256)
void transpose_cvt_kernel(const float* __restrict__ W, __bf16* __restrict__ Wt) {
  __shared__ float tile[32][33];
  int tx = threadIdx.x, ty = threadIdx.y;          // blockDim = (32, 8)
  int x0 = blockIdx.x * 32, y0 = blockIdx.y * 32;  // x: k, y: n
#pragma unroll
  for (int j = 0; j < 32; j += 8)
    tile[ty + j][tx] = W[(size_t)(y0 + ty + j) * DIM + x0 + tx];
  __syncthreads();
#pragma unroll
  for (int j = 0; j < 32; j += 8)
    Wt[(size_t)(x0 + ty + j) * DIM + y0 + tx] = f2b(tile[tx][ty + j]);
}

// ---------- GEMM: C[m][n] = sum_k A[m][k] * Wt[k][n] + bias[n] ----------
// One wave per 32x64 tile; 8 accumulators; K stepped by 32 with register
// double-buffering so fragment loads overlap the WMMA batch.
__global__ __launch_bounds__(32)
void gemm_bf16_kernel(const __bf16* __restrict__ A, const __bf16* __restrict__ Bt,
                      const float* __restrict__ bias, float* __restrict__ Cf,
                      __bf16* __restrict__ Cb, int Kd, int Nd, int out_bf16) {
  int n0 = blockIdx.x * 64;
  int m0 = blockIdx.y * 32;
  int lane = threadIdx.x;
  v8f acc[2][4] = {};

  v16bf a0[2], a1[2], b0[4], b1[4];
#pragma unroll
  for (int i = 0; i < 2; ++i) a0[i] = load_a_frag(A, Kd, m0 + i * 16, 0);
#pragma unroll
  for (int c = 0; c < 4; ++c) b0[c] = load_b_frag(Bt, Nd, 0, n0 + c * 16);

  for (int k0 = 0; k0 < Kd; k0 += 64) {  // Kd is a multiple of 64
    // issue loads for k0+32 (set 1)
#pragma unroll
    for (int i = 0; i < 2; ++i) a1[i] = load_a_frag(A, Kd, m0 + i * 16, k0 + 32);
#pragma unroll
    for (int c = 0; c < 4; ++c) b1[c] = load_b_frag(Bt, Nd, k0 + 32, n0 + c * 16);
    if (k0 + 128 < Kd) {
      __builtin_prefetch(A  + (size_t)(m0 + (lane & 15)) * Kd + k0 + 128, 0, 3);
      __builtin_prefetch(Bt + (size_t)(k0 + 128 + lane) * Nd + n0, 0, 3);
    }
    // compute on set 0
#pragma unroll
    for (int i = 0; i < 2; ++i)
#pragma unroll
      for (int c = 0; c < 4; ++c)
        acc[i][c] = __builtin_amdgcn_wmma_f32_16x16x32_bf16(false, a0[i], false, b0[c],
                                                            (short)0, acc[i][c], false, false);
    // issue loads for k0+64 (set 0)
    if (k0 + 64 < Kd) {
#pragma unroll
      for (int i = 0; i < 2; ++i) a0[i] = load_a_frag(A, Kd, m0 + i * 16, k0 + 64);
#pragma unroll
      for (int c = 0; c < 4; ++c) b0[c] = load_b_frag(Bt, Nd, k0 + 64, n0 + c * 16);
    }
    // compute on set 1
#pragma unroll
    for (int i = 0; i < 2; ++i)
#pragma unroll
      for (int c = 0; c < 4; ++c)
        acc[i][c] = __builtin_amdgcn_wmma_f32_16x16x32_bf16(false, a1[i], false, b1[c],
                                                            (short)0, acc[i][c], false, false);
  }

  int col = lane & 15, half = lane >> 4;
#pragma unroll
  for (int i = 0; i < 2; ++i)
#pragma unroll
    for (int c = 0; c < 4; ++c) {
      int nn = n0 + c * 16 + col;
      float bv = bias[nn];
#pragma unroll
      for (int r = 0; r < 8; ++r) {
        int mm = m0 + i * 16 + r + 8 * half;
        float v = acc[i][c][r] + bv;
        if (out_bf16) Cb[(size_t)mm * Nd + nn] = f2b(v);
        else          Cf[(size_t)mm * Nd + nn] = v;
      }
    }
}

// ---------- per-(token, head) RMSNorm over HD=64, f32 in -> bf16 out ----------
__global__ __launch_bounds__(32)
void rmsnorm_kernel(const float* __restrict__ x, const float* __restrict__ w,
                    __bf16* __restrict__ out) {
  size_t base = (size_t)blockIdx.x * HDIM;
  int lane = threadIdx.x;
  float x0 = x[base + lane], x1 = x[base + 32 + lane];
  float ss = x0 * x0 + x1 * x1;
#pragma unroll
  for (int o = 16; o >= 1; o >>= 1) ss += __shfl_xor(ss, o, 32);
  float inv = rsqrtf(ss * (1.0f / 64.0f) + 1e-5f);
  out[base + lane]      = f2b(w[lane]      * x0 * inv);
  out[base + 32 + lane] = f2b(w[lane + 32] * x1 * inv);
}

// ---------- flash attention: one wave per (head, 16-query tile) ----------
// S^T = K_tile x Q^T via WMMA; the S^T C-tile layout is exactly the A-fragment
// layout for the P(16x32) x V(32x64) WMMAs -> no shuffles after softmax.
// K/V tiles stream through LDS: TDM double-buffered when available, else staged
// with plain global loads.
__global__ __launch_bounds__(32)
void attention_kernel(const __bf16* __restrict__ Q, const __bf16* __restrict__ K,
                      const __bf16* __restrict__ V, __bf16* __restrict__ Obuf) {
  __shared__ __align__(32) __bf16 q_lds[16 * HDIM];
  __shared__ __align__(32) __bf16 k_lds[2][32 * HDIM];
  __shared__ __align__(32) __bf16 v_lds[2][32 * HDIM];
  int q0 = blockIdx.x * 16;
  int h  = blockIdx.y;
  int lane = threadIdx.x;
  int col = lane & 15, half = lane >> 4;

  // Stage the 16x64 Q tile into LDS.
  {
    int row = lane >> 1, off = (lane & 1) * 32;
    const __bf16* src = Q + (size_t)(q0 + row) * DIM + h * HDIM + off;
    __bf16* dst = &q_lds[row * HDIM + off];
#pragma unroll
    for (int t = 0; t < 4; ++t)
      *(v8bf*)(dst + t * 8) = *(const v8bf*)(src + t * 8);
  }
  __syncthreads();

  // Q^T B-fragments (two 32-dim K-halves): lane = query (lane&15), elems = dims in table order.
  v16bf bq[2];
#pragma unroll
  for (int kk = 0; kk < 2; ++kk) {
    int d = kk * 32 + 8 * half;
#pragma unroll
    for (int e = 0; e < 8; ++e) {
      bq[kk][e]     = q_lds[col * HDIM + d + e];
      bq[kk][e + 8] = q_lds[col * HDIM + d + 16 + e];
    }
  }

#if ATTN_USE_TDM
  // Preload the first K/V tiles into buffer 0 via the Tensor Data Mover.
  tdm_load_tile_2d(&k_lds[0][0], K + h * HDIM, HDIM, 32, DIM, N_TOK, DIM);
  tdm_load_tile_2d(&v_lds[0][0], V + h * HDIM, HDIM, 32, DIM, N_TOK, DIM);
#endif

  v8f o[4] = {};                 // O accumulator: 16 queries x 64 dims
  float m = -3.0e38f, l = 0.0f;  // online-softmax stats for query (lane&15)
  int buf = 0;

  for (int j0 = 0; j0 < N_TOK; j0 += 32) {
#if ATTN_USE_TDM
    if (j0 + 32 < N_TOK) {
      // Issue DMA for the next tile pair, then wait until the current pair landed.
      tdm_load_tile_2d(&k_lds[buf ^ 1][0], K + (size_t)(j0 + 32) * DIM + h * HDIM,
                       HDIM, 32, DIM, N_TOK, DIM);
      tdm_load_tile_2d(&v_lds[buf ^ 1][0], V + (size_t)(j0 + 32) * DIM + h * HDIM,
                       HDIM, 32, DIM, N_TOK, DIM);
      __builtin_amdgcn_s_wait_tensorcnt(2);
    } else {
      __builtin_amdgcn_s_wait_tensorcnt(0);
    }
#else
    // Stage current 32x64 K/V tiles into LDS (lane = key row).
    {
      const __bf16* ksrc = K + (size_t)(j0 + lane) * DIM + h * HDIM;
      const __bf16* vsrc = V + (size_t)(j0 + lane) * DIM + h * HDIM;
      if (j0 + 32 < N_TOK) {
        __builtin_prefetch(K + (size_t)(j0 + 32 + lane) * DIM + h * HDIM, 0, 3);
        __builtin_prefetch(V + (size_t)(j0 + 32 + lane) * DIM + h * HDIM, 0, 3);
      }
#pragma unroll
      for (int t = 0; t < 4; ++t) {
        *(v8bf*)&k_lds[buf][lane * HDIM + t * 8] = *(const v8bf*)(ksrc + t * 8);
        *(v8bf*)&v_lds[buf][lane * HDIM + t * 8] = *(const v8bf*)(vsrc + t * 8);
      }
    }
    __syncthreads();
#endif

    // S^T tiles: st[t] covers keys j0+16t..+15 (M) x 16 queries (N), K-dim=64.
    v8f st[2] = {};
#pragma unroll
    for (int t = 0; t < 2; ++t) {
#pragma unroll
      for (int kk = 0; kk < 2; ++kk) {
        const __bf16* p = &k_lds[buf][(t * 16 + col) * HDIM + kk * 32 + 8 * half];
        v8bf lo = *(const v8bf*)p;
        v8bf hi = *(const v8bf*)(p + 16);
        v16bf aK;
#pragma unroll
        for (int i = 0; i < 8; ++i) { aK[i] = lo[i]; aK[i + 8] = hi[i]; }
        st[t] = __builtin_amdgcn_wmma_f32_16x16x32_bf16(false, aK, false, bq[kk],
                                                        (short)0, st[t], false, false);
      }
    }

    // Online softmax. Lane holds, for query col, keys (8*half + r) and (16 + 8*half + r).
    float tm = st[0][0];
#pragma unroll
    for (int r = 0; r < 8; ++r) { tm = fmaxf(tm, st[0][r]); tm = fmaxf(tm, st[1][r]); }
    tm = fmaxf(tm, __shfl_xor(tm, 16, 32));
    float newm = fmaxf(m, tm);
    float alpha = __expf(m - newm);
    float rsum = 0.0f;
    v16bf pf;  // probabilities in exact 16x32 A-fragment layout
#pragma unroll
    for (int r = 0; r < 8; ++r) {
      float p0 = __expf(st[0][r] - newm);
      float p1 = __expf(st[1][r] - newm);
      rsum += p0 + p1;
      pf[r]     = f2b(p0);
      pf[r + 8] = f2b(p1);
    }
    rsum += __shfl_xor(rsum, 16, 32);
    l = l * alpha + rsum;
    m = newm;

    // Rescale O by per-row alpha (row of O-tile = query r + 8*half).
    float ar[8];
#pragma unroll
    for (int r = 0; r < 8; ++r) ar[r] = __shfl(alpha, r + 8 * half, 32);
#pragma unroll
    for (int c = 0; c < 4; ++c)
#pragma unroll
      for (int r = 0; r < 8; ++r) o[c][r] *= ar[r];

    // O += P x V: B-frag lane = head dim (lane&15), elems = keys in table order.
#pragma unroll
    for (int c = 0; c < 4; ++c) {
      v16bf bV;
#pragma unroll
      for (int s = 0; s < 8; ++s) {
        bV[s]     = v_lds[buf][(8 * half + s) * HDIM + c * 16 + col];
        bV[s + 8] = v_lds[buf][(16 + 8 * half + s) * HDIM + c * 16 + col];
      }
      o[c] = __builtin_amdgcn_wmma_f32_16x16x32_bf16(false, pf, false, bV,
                                                     (short)0, o[c], false, false);
    }
    buf ^= 1;
  }

  float lr[8];
#pragma unroll
  for (int r = 0; r < 8; ++r) lr[r] = __shfl(l, r + 8 * half, 32);
#pragma unroll
  for (int c = 0; c < 4; ++c) {
#pragma unroll
    for (int r = 0; r < 8; ++r) {
      size_t idx = (size_t)(q0 + r + 8 * half) * DIM + h * HDIM + c * 16 + col;
      Obuf[idx] = f2b(o[c][r] / lr[r]);
    }
  }
}

// ---------- host launcher ----------
extern "C" void kernel_launch(void* const* d_in, const int* in_sizes, int n_in,
                              void* d_out, int out_size, void* d_ws, size_t ws_size,
                              hipStream_t stream) {
  (void)in_sizes; (void)n_in; (void)out_size; (void)ws_size;
  const float* query = (const float*)d_in[0];
  const float* key   = (const float*)d_in[1];
  const float* value = (const float*)d_in[2];
  const float* Wq = (const float*)d_in[3];  const float* bq = (const float*)d_in[4];
  const float* Wk = (const float*)d_in[5];  const float* bk = (const float*)d_in[6];
  const float* Wv = (const float*)d_in[7];  const float* bv = (const float*)d_in[8];
  const float* Wo = (const float*)d_in[9];  const float* bo = (const float*)d_in[10];
  const float* qnw = (const float*)d_in[11];
  const float* knw = (const float*)d_in[12];

  char* ws = (char*)d_ws;
  size_t off = 0;
  auto take = [&](size_t nbytes) -> void* {
    void* p = ws + off;
    off += (nbytes + 255) & ~(size_t)255;
    return p;
  };
  const size_t ND = (size_t)N_TOK * DIM;
  const size_t DD = (size_t)DIM * DIM;

  __bf16* aq  = (__bf16*)take(ND * 2);
  __bf16* ak  = (__bf16*)take(ND * 2);
  __bf16* av  = (__bf16*)take(ND * 2);
  __bf16* wqt = (__bf16*)take(DD * 2);
  __bf16* wkt = (__bf16*)take(DD * 2);
  __bf16* wvt = (__bf16*)take(DD * 2);
  __bf16* wot = (__bf16*)take(DD * 2);
  float*  qf  = (float*) take(ND * 4);
  float*  kf  = (float*) take(ND * 4);
  __bf16* qb  = (__bf16*)take(ND * 2);
  __bf16* kb  = (__bf16*)take(ND * 2);
  __bf16* vb  = (__bf16*)take(ND * 2);
  __bf16* atb = (__bf16*)take(ND * 2);

  // 1) activations f32 -> bf16
  {
    int n = (int)ND, blocks = (n + 255) / 256;
    cvt_bf16_kernel<<<blocks, 256, 0, stream>>>(query, aq, n);
    cvt_bf16_kernel<<<blocks, 256, 0, stream>>>(key,   ak, n);
    cvt_bf16_kernel<<<blocks, 256, 0, stream>>>(value, av, n);
  }
  // 2) weights: transpose + convert -> Wt[k][n] bf16
  {
    dim3 g(DIM / 32, DIM / 32), b(32, 8);
    transpose_cvt_kernel<<<g, b, 0, stream>>>(Wq, wqt);
    transpose_cvt_kernel<<<g, b, 0, stream>>>(Wk, wkt);
    transpose_cvt_kernel<<<g, b, 0, stream>>>(Wv, wvt);
    transpose_cvt_kernel<<<g, b, 0, stream>>>(Wo, wot);
  }
  // 3) projections (WMMA GEMMs, 32x64 tiles, double-buffered)
  {
    dim3 g(DIM / 64, N_TOK / 32);
    gemm_bf16_kernel<<<g, 32, 0, stream>>>(aq, wqt, bq, qf, nullptr, DIM, DIM, 0);
    gemm_bf16_kernel<<<g, 32, 0, stream>>>(ak, wkt, bk, kf, nullptr, DIM, DIM, 0);
    gemm_bf16_kernel<<<g, 32, 0, stream>>>(av, wvt, bv, nullptr, vb, DIM, DIM, 1);
  }
  // 4) QK RMSNorm (per token x head, HD=64) -> bf16
  {
    int blocks = N_TOK * NHEAD;
    rmsnorm_kernel<<<blocks, 32, 0, stream>>>(qf, qnw, qb);
    rmsnorm_kernel<<<blocks, 32, 0, stream>>>(kf, knw, kb);
  }
  // 5) flash attention (WMMA + TDM-streamed K/V) -> attn bf16 [N, H*HD]
  {
    dim3 g(N_TOK / 16, NHEAD);
    attention_kernel<<<g, 32, 0, stream>>>(qb, kb, vb, atb);
  }
  // 6) output projection -> f32 d_out
  {
    dim3 g(DIM / 64, N_TOK / 32);
    gemm_bf16_kernel<<<g, 32, 0, stream>>>(atb, wot, bo, (float*)d_out, nullptr, DIM, DIM, 0);
  }
}
